// RefinementLayer3_48945447306025
// MI455X (gfx1250) — matrix operand order repaired and checked by reading, not compiled, not run on hardware
//
#include <hip/hip_runtime.h>

// Problem constants (B=2, L=128, D=768, H=256, C=4)
#define BB 2
#define LL 128
#define DD 768
#define HH 256
#define CC 4

typedef _Float16 f16;
typedef __attribute__((ext_vector_type(16))) _Float16 v16h;
typedef __attribute__((ext_vector_type(8)))  float    v8f;

static __device__ __forceinline__ v8f wmma_f16(v16h a, v16h b, v8f c) {
  // (neg_a, A, neg_b, B, c_mod, C, reuse_a, reuse_b) -> v_wmma_f32_16x16x32_f16
  return __builtin_amdgcn_wmma_f32_16x16x32_f16(false, a, false, b, (short)0, c,
                                                false, false);
}

static __device__ __forceinline__ float fast_tanh(float x) {
#if __has_builtin(__builtin_amdgcn_tanhf)
  return __builtin_amdgcn_tanhf(x);   // v_tanh_f32 (TRANS pipe)
#else
  return tanhf(x);
#endif
}

// ---- f16 fragment loaders (wave32, contiguous k order: k = 16*(lane/16) + s) ----
// A tile 16(M) x 32(K) from row-major [M x K] f16 (row stride lda).
// lanes 0-15: K = k0..k0+15; lanes 16-31: K = k0+16..k0+31.
static __device__ __forceinline__ v16h load_a_rm(const f16* __restrict__ A,
                                                 int lda, int m0, int k0,
                                                 int lane) {
  const int row = m0 + (lane & 15);
  const int kb  = k0 + ((lane & 16) ? 16 : 0);
  const f16* p = A + row * lda + kb;
  v16h r;
#pragma unroll
  for (int i = 0; i < 8; ++i) r[i] = p[i];
#pragma unroll
  for (int i = 0; i < 8; ++i) r[8 + i] = p[8 + i];
  return r;
}

// B tile 32(K) x 16(N) from row-major [K x N] f16 (row stride ldb).
// lane n in 0-15: K = k0..k0+15; lanes 16-31: K = k0+16..k0+31.
static __device__ __forceinline__ v16h load_b_kn(const f16* __restrict__ B,
                                                 int ldb, int k0, int n0,
                                                 int lane) {
  const int col = n0 + (lane & 15);
  const int kb  = k0 + ((lane & 16) ? 16 : 0);
  const f16* p = B + kb * ldb + col;
  v16h r;
#pragma unroll
  for (int i = 0; i < 16; ++i) r[i] = p[i * ldb];
  return r;
}

// B tile 32(K) x 16(N) where f16 memory holds [N x K] row-major (B = W^T).
static __device__ __forceinline__ v16h load_b_nk(const f16* __restrict__ W,
                                                 int ldw, int k0, int n0,
                                                 int lane) {
  const int col = n0 + (lane & 15);
  const int kb  = k0 + ((lane & 16) ? 16 : 0);
  const f16* p = W + col * ldw + kb;
  v16h r;
#pragma unroll
  for (int i = 0; i < 16; ++i) r[i] = p[i];
  return r;
}

// ---------------------------------------------------------------------------
// Pre-pass: one-shot f32 -> f16 conversion of GEMM operands.
// ---------------------------------------------------------------------------
__global__ void cvt_f32_to_f16(const float* __restrict__ src,
                               f16* __restrict__ dst, int n) {
  const int i = blockIdx.x * blockDim.x + threadIdx.x;
  if (i < n) dst[i] = (f16)src[i];
}

// ---------------------------------------------------------------------------
// Stage A: h_p = tanh(seq @ W_prd^T + b_prd)  -> hcatH[:, 0:256]   (f16)
//          h_a =       seq @ U_arg^T + b_arg  -> hAH (raw f16), taH = tanh (f16)
// seq: (B*L=256) x 768. One wave per 16x16 output tile.
// ---------------------------------------------------------------------------
__global__ void stageA_proj(const f16* __restrict__ seqH,
                            const f16* __restrict__ WprdH,
                            const float* __restrict__ b_prd,
                            const f16* __restrict__ UargH,
                            const float* __restrict__ b_arg,
                            f16* __restrict__ hcatH,   // 256 x 1280
                            f16* __restrict__ hAH,     // 256 x 1024
                            f16* __restrict__ taH) {   // 256 x 1024
  const int lane = threadIdx.x;
  const int m0   = blockIdx.x * 16;                       // row tile in [0,256)
  const int n0g  = (blockIdx.y * 4 + threadIdx.y) * 16;   // col tile in [0,1280)
  const bool isP = (n0g < HH);
  const int  n0  = isP ? n0g : (n0g - HH);
  const f16*   Wt   = isP ? WprdH : UargH;  // (N x K) row-major, ld = 768
  const float* bias = isP ? b_prd : b_arg;

  v8f acc = {};
#pragma unroll 4
  for (int k0 = 0; k0 < DD; k0 += 32) {
    v16h a = load_a_rm(seqH, DD, m0, k0, lane);
    v16h b = load_b_nk(Wt, DD, k0, n0, lane);
    acc = wmma_f16(a, b, acc);
  }
  const int n  = n0 + (lane & 15);
  const int mb = m0 + ((lane & 16) ? 8 : 0);
  const float bv = bias[n];
#pragma unroll
  for (int r = 0; r < 8; ++r) {
    const int row = mb + r;
    const float v = acc[r] + bv;
    if (isP) {
      hcatH[row * (HH + CC * HH) + n] = (f16)fast_tanh(v);
    } else {
      hAH[row * (CC * HH) + n] = (f16)v;
      taH[row * (CC * HH) + n] = (f16)fast_tanh(v);
    }
  }
}

// ---------------------------------------------------------------------------
// Stage B: ws[b,p,c,h] = sum_a base[b,p,c,a] * hA[b,a,c,h];
//          hcatH[., 256 + c*H + h] = tanh(ws).  Per (b,c): 128 x 128 x 256.
// ---------------------------------------------------------------------------
__global__ void stageB_ws(const f16* __restrict__ baseH,
                          const f16* __restrict__ hAH,
                          f16* __restrict__ hcatH) {
  const int lane = threadIdx.x;
  const int m0 = blockIdx.x * 16;                      // p tile in [0,128)
  const int n0 = (blockIdx.y * 4 + threadIdx.y) * 16;  // h tile in [0,256)
  const int z = blockIdx.z;
  const int b = z >> 2, c = z & 3;

  const f16* A  = baseH + ((size_t)b * LL * CC + c) * LL;     // lda = 512
  const f16* Bm = hAH + (size_t)b * LL * (CC * HH) + c * HH;  // ldb = 1024

  v8f acc = {};
#pragma unroll
  for (int k0 = 0; k0 < LL; k0 += 32) {
    v16h a = load_a_rm(A, CC * LL, m0, k0, lane);
    v16h bb = load_b_kn(Bm, CC * HH, k0, n0, lane);
    acc = wmma_f16(a, bb, acc);
  }
  const int n  = n0 + (lane & 15);
  const int mb = m0 + ((lane & 16) ? 8 : 0);
  f16* O = hcatH + (size_t)b * LL * (HH + CC * HH) + HH + c * HH;
#pragma unroll
  for (int r = 0; r < 8; ++r)
    O[(mb + r) * (HH + CC * HH) + n] = (f16)fast_tanh(acc[r]);
}

// ---------------------------------------------------------------------------
// Stage C-X: X[b,p,c,d] = [tp|tr](b,p,:) @ [W1[c];W3[c]] + b_mid[c,d]  (f32)
// K = 1280 (k<256 -> W_mid row k (W1); k>=256 -> W_mid row k+512 (W3)).
// ---------------------------------------------------------------------------
__global__ void stageC_X(const f16* __restrict__ hcatH,
                         const f16* __restrict__ WmidH,
                         const float* __restrict__ b_mid,
                         float* __restrict__ X) {  // (B*L) x (C*H), [b,p,c,d]
  const int lane = threadIdx.x;
  const int m0 = blockIdx.x * 16;                      // p tile
  const int n0 = (blockIdx.y * 4 + threadIdx.y) * 16;  // d tile
  const int z = blockIdx.z;
  const int b = z >> 2, c = z & 3;

  const f16* A = hcatH + (size_t)b * LL * (HH + CC * HH);     // lda = 1280
  const f16* W = WmidH + (size_t)c * ((CC + 2) * HH) * HH;    // rows x 256

  v8f acc = {};
#pragma unroll 4
  for (int k0 = 0; k0 < (HH + CC * HH); k0 += 32) {
    const int krow0 = (k0 < HH) ? k0 : (k0 + 2 * HH);  // skip W2 rows
    v16h a = load_a_rm(A, HH + CC * HH, m0, k0, lane);
    v16h bb = load_b_kn(W, HH, krow0, n0, lane);
    acc = wmma_f16(a, bb, acc);
  }
  const int n  = n0 + (lane & 15);
  const int mb = m0 + ((lane & 16) ? 8 : 0);
  const float bv = b_mid[c * HH + n];
  float* Xp = X + (size_t)b * LL * (CC * HH) + c * HH;
#pragma unroll
  for (int r = 0; r < 8; ++r)
    Xp[(mb + r) * (CC * HH) + n] = acc[r] + bv;
}

// ---------------------------------------------------------------------------
// Stage C-Y: Y[b,a,c,d] = ta(b,a,c,:) @ W2[c]; stored TRANSPOSED as Yt[b,c,d,a]
// (f32) so stage D (lane == a) gets coalesced loads.
// ---------------------------------------------------------------------------
__global__ void stageC_Y(const f16* __restrict__ taH,
                         const f16* __restrict__ WmidH,
                         float* __restrict__ Yt) {  // [b,c,d,a] = (B*C*H) x L
  const int lane = threadIdx.x;
  const int m0 = blockIdx.x * 16;                      // a tile in [0,128)
  const int n0 = (blockIdx.y * 4 + threadIdx.y) * 16;  // d tile in [0,256)
  const int z = blockIdx.z;
  const int b = z >> 2, c = z & 3;

  const f16* A = taH + (size_t)b * LL * (CC * HH) + c * HH;        // lda = 1024
  const f16* W = WmidH + ((size_t)c * ((CC + 2) * HH) + HH) * HH;  // W2 rows

  v8f acc = {};
#pragma unroll
  for (int k0 = 0; k0 < HH; k0 += 32) {
    v16h a = load_a_rm(A, CC * HH, m0, k0, lane);
    v16h bb = load_b_kn(W, HH, k0, n0, lane);
    acc = wmma_f16(a, bb, acc);
  }
  const int n  = n0 + (lane & 15);            // d
  const int mb = m0 + ((lane & 16) ? 8 : 0);  // a base
  float* Yp = Yt + (((size_t)(b * CC + c)) * HH + n) * LL;
#pragma unroll
  for (int r = 0; r < 8; ++r)
    Yp[mb + r] = acc[r];
}

// ---------------------------------------------------------------------------
// Stage D: out[b,p,c,a] = sum_d tanh(X[b,p,c,d] + Yt[b,c,d,a]) * w_out[d]
// One thread per output; lanes vary over `a` -> X loads wave-uniform, Yt loads
// coalesced. Hardware v_tanh_f32 (TRANS) bound.
// ---------------------------------------------------------------------------
__global__ void stageD_pair(const float* __restrict__ X,
                            const float* __restrict__ Yt,
                            const float* __restrict__ w_out,
                            float* __restrict__ out) {
  __shared__ float w[HH];
  for (int i = threadIdx.x; i < HH; i += blockDim.x) w[i] = w_out[i];
  __syncthreads();

  const int idx = blockIdx.x * blockDim.x + threadIdx.x;  // < B*L*C*L = 131072
  const int a = idx & (LL - 1);
  const int c = (idx >> 7) & (CC - 1);
  const int p = (idx >> 9) & (LL - 1);
  const int b = idx >> 16;

  const float* xp = X + (((size_t)(b * LL + p)) * CC + c) * HH;
  const float* yp = Yt + ((size_t)(b * CC + c)) * HH * LL + a;

  float acc = 0.0f;
#pragma unroll 4
  for (int d = 0; d < HH; ++d)
    acc = fmaf(fast_tanh(xp[d] + yp[(size_t)d * LL]), w[d], acc);

  out[idx] = acc;
}

// ---------------------------------------------------------------------------
extern "C" void kernel_launch(void* const* d_in, const int* in_sizes, int n_in,
                              void* d_out, int out_size, void* d_ws, size_t ws_size,
                              hipStream_t stream) {
  const float* seq    = (const float*)d_in[0];  // (2,128,768)
  const float* base   = (const float*)d_in[1];  // (2,128,4,128)
  const float* W_prd  = (const float*)d_in[2];  // (256,768)
  const float* b_prd  = (const float*)d_in[3];  // (256,)
  const float* U_arg  = (const float*)d_in[4];  // (1024,768)
  const float* b_arg  = (const float*)d_in[5];  // (1024,)
  const float* W_mid  = (const float*)d_in[6];  // (4,1536,256)
  const float* b_mid  = (const float*)d_in[7];  // (4,256)
  const float* w_out  = (const float*)d_in[8];  // (256,)
  float* out = (float*)d_out;                   // (2,128,4,128)

  // Workspace layout: f32 region first (alignment), then f16 region.
  float* X  = (float*)d_ws;                      // 256*1024 f32  [b,p,c,d]
  float* Yt = X + 256 * 1024;                    // 2*4*256*128 f32 [b,c,d,a]
  f16* seqH  = (f16*)(Yt + 256 * 1024);
  f16* WprdH = seqH + 256 * 768;
  f16* UargH = WprdH + 256 * 768;
  f16* WmidH = UargH + 1024 * 768;
  f16* baseH = WmidH + CC * 1536 * 256;
  f16* hcatH = baseH + BB * LL * CC * LL;        // 256 x 1280
  f16* hAH   = hcatH + 256 * 1280;               // 256 x 1024
  f16* taH   = hAH + 256 * 1024;                 // 256 x 1024

  // Pre-pass: one-shot conversions of all GEMM operands to f16.
  {
    const int nt = 256;
    cvt_f32_to_f16<<<(256 * 768 + nt - 1) / nt, nt, 0, stream>>>(seq, seqH, 256 * 768);
    cvt_f32_to_f16<<<(256 * 768 + nt - 1) / nt, nt, 0, stream>>>(W_prd, WprdH, 256 * 768);
    cvt_f32_to_f16<<<(1024 * 768 + nt - 1) / nt, nt, 0, stream>>>(U_arg, UargH, 1024 * 768);
    cvt_f32_to_f16<<<(CC * 1536 * 256 + nt - 1) / nt, nt, 0, stream>>>(W_mid, WmidH, CC * 1536 * 256);
    cvt_f32_to_f16<<<(BB * LL * CC * LL + nt - 1) / nt, nt, 0, stream>>>(base, baseH, BB * LL * CC * LL);
  }

  dim3 wblk(32, 4, 1);  // 4 wave32s per block, one 16x16 tile each

  // Stage A: projections. M=256 rows, N=1280 cols total.
  stageA_proj<<<dim3(16, 20, 1), wblk, 0, stream>>>(seqH, WprdH, b_prd, UargH,
                                                    b_arg, hcatH, hAH, taH);
  // Stage B: ws GEMMs per (b,c). M=128, N=256, K=128.
  stageB_ws<<<dim3(8, 4, BB * CC), wblk, 0, stream>>>(baseH, hAH, hcatH);
  // Stage C: X (K=1280) and Y (K=256) per (b,c). M=128, N=256.
  stageC_X<<<dim3(8, 4, BB * CC), wblk, 0, stream>>>(hcatH, WmidH, b_mid, X);
  stageC_Y<<<dim3(8, 4, BB * CC), wblk, 0, stream>>>(taH, WmidH, Yt);
  // Stage D: pairwise tanh reduction. 131072 outputs.
  stageD_pair<<<dim3((BB * LL * CC * LL) / 256, 1, 1), dim3(256, 1, 1), 0,
                stream>>>(X, Yt, w_out, out);
}
